// ModelGCNAttn3h_77884936945817
// MI455X (gfx1250) — compile-verified, matrix operand order checked by reading, not run on hardware
//
#include <hip/hip_runtime.h>
#include <hip/hip_bf16.h>

// ---------------------------------------------------------------------------
// ModelGCNAttn3h for MI455X (gfx1250, wave32, WMMA).
// - All dense contractions: v_wmma_f32_16x16x32_f16 (fp32 storage, f16
//   convert-on-load via float4 vector loads, fp32 accumulate).
// - Attention scores+softmax fused in one kernel: the 268 MB w3h tensor is
//   written to HBM exactly once (roofline floor ~11.5us @ 23.3 TB/s).
// - Boundary tiles use clamp-address + value-select loads (no exec-masked
//   per-element loads -> loads pipeline under a single s_wait).
// - Sparse GCN aggregation via fp32 global atomics.
// ---------------------------------------------------------------------------

typedef __attribute__((ext_vector_type(16))) _Float16 v16h;
typedef __attribute__((ext_vector_type(8)))  float    v8f;

#define HBLK 128

// ---- fragment loaders ------------------------------------------------------
// A/B(trans) fragment: lane holds one row; halves 0..7 <- [kA..kA+7],
// halves 8..15 <- [kB..kB+7] (ISA 7.12.2 16-bit A layout, kA=8*hi, kB=16+8*hi).
__device__ __forceinline__ v16h ld_vec(const float* __restrict__ p, int kA, int kB) {
    float4 a0 = *(const float4*)(p + kA);
    float4 a1 = *(const float4*)(p + kA + 4);
    float4 b0 = *(const float4*)(p + kB);
    float4 b1 = *(const float4*)(p + kB + 4);
    v16h f;
    f[0]=(_Float16)a0.x; f[1]=(_Float16)a0.y; f[2]=(_Float16)a0.z;  f[3]=(_Float16)a0.w;
    f[4]=(_Float16)a1.x; f[5]=(_Float16)a1.y; f[6]=(_Float16)a1.z;  f[7]=(_Float16)a1.w;
    f[8]=(_Float16)b0.x; f[9]=(_Float16)b0.y; f[10]=(_Float16)b0.z; f[11]=(_Float16)b0.w;
    f[12]=(_Float16)b1.x;f[13]=(_Float16)b1.y;f[14]=(_Float16)b1.z; f[15]=(_Float16)b1.w;
    return f;
}

// clamped row-major loader: unconditional loads from clamped address, then select
__device__ __forceinline__ v16h ld_row_clamped(const float* __restrict__ base, long long ld,
                                               int row, int rmax, int kA, int kB, int K) {
    int  rc = row < rmax ? row : rmax - 1;
    bool rv = row < rmax;
    const float* p = base + (long long)rc * ld;
    v16h f;
#pragma unroll
    for (int i = 0; i < 8; ++i) {
        int k1 = kA + i, k2 = kB + i;
        float v1 = p[k1 < K ? k1 : 0];
        float v2 = p[k2 < K ? k2 : 0];
        f[i]     = (rv && k1 < K) ? (_Float16)v1 : (_Float16)0.0f;
        f[i + 8] = (rv && k2 < K) ? (_Float16)v2 : (_Float16)0.0f;
    }
    return f;
}

// clamped column loader for B stored (K,N): element B[k][col], stride ld
__device__ __forceinline__ v16h ld_col_clamped(const float* __restrict__ base, long long ld,
                                               int col, int cmax, int kA, int kB, int K) {
    int  cc = col < cmax ? col : cmax - 1;
    bool cv = col < cmax;
    v16h f;
#pragma unroll
    for (int i = 0; i < 8; ++i) {
        int k1 = kA + i, k2 = kB + i;
        float v1 = base[(long long)(k1 < K ? k1 : 0) * ld + cc];
        float v2 = base[(long long)(k2 < K ? k2 : 0) * ld + cc];
        f[i]     = (cv && k1 < K) ? (_Float16)v1 : (_Float16)0.0f;
        f[i + 8] = (cv && k2 < K) ? (_Float16)v2 : (_Float16)0.0f;
    }
    return f;
}

// ---------------------------------------------------------------------------
// Generic batched WMMA GEMM: C = act(alpha * A x B' + bias)
//   bTrans=1: B stored (N,K) row-major (linear weights), vectorized path
//   bTrans=0: B stored (K,N) row-major (strided loads)
// ---------------------------------------------------------------------------
__global__ void __launch_bounds__(128)
gemm_wmma_f16(const float* __restrict__ A, int lda, long long sAb, long long sAh,
              const float* __restrict__ Bm, int ldb, long long sBb, long long sBh,
              float* __restrict__ C, int ldc, long long sCb, long long sCh,
              const float* __restrict__ bias,
              int M, int N, int K, int nH, int bTrans, float alpha, int act)
{
    int z  = blockIdx.z;
    int bb = z / nH, hh = z % nH;
    const float* Ab = A  + bb * sAb + hh * sAh;
    const float* Bb = Bm + bb * sBb + hh * sBh;
    float*       Cb = C  + bb * sCb + hh * sCh;

    int tn = blockIdx.x * 4 + threadIdx.y;
    int tm = blockIdx.y;
    int n0 = tn * 16, m0 = tm * 16;
    if (n0 >= N || m0 >= M) return;          // wave-uniform

    int lane = threadIdx.x;
    int r16  = lane & 15;
    int hi   = lane >> 4;
    int kA0  = 8 * hi, kB0 = 16 + 8 * hi;

    int arow = m0 + r16, brow = n0 + r16;
    bool interior = (m0 + 16 <= M) && (n0 + 16 <= N);
    int K32 = K & ~31;

    v8f acc = {};
    if (interior) {
        const float* ap = Ab + (long long)arow * lda;
        if (bTrans) {
            const float* bp = Bb + (long long)brow * ldb;
            for (int k0 = 0; k0 < K32; k0 += 32) {
                v16h af = ld_vec(ap + k0, kA0, kB0);
                v16h bf = ld_vec(bp + k0, kA0, kB0);
                acc = __builtin_amdgcn_wmma_f32_16x16x32_f16(false, af, false, bf,
                                                             (short)0, acc, false, false);
            }
        } else {
            for (int k0 = 0; k0 < K32; k0 += 32) {
                v16h af = ld_vec(ap + k0, kA0, kB0);
                v16h bf;
#pragma unroll
                for (int i = 0; i < 8; ++i) {
                    bf[i]     = (_Float16)Bb[(long long)(k0 + kA0 + i) * ldb + brow];
                    bf[i + 8] = (_Float16)Bb[(long long)(k0 + kB0 + i) * ldb + brow];
                }
                acc = __builtin_amdgcn_wmma_f32_16x16x32_f16(false, af, false, bf,
                                                             (short)0, acc, false, false);
            }
        }
    } else {
        for (int k0 = 0; k0 < K32; k0 += 32) {
            v16h af = ld_row_clamped(Ab, lda, arow, M, k0 + kA0, k0 + kB0, K);
            v16h bf = bTrans ? ld_row_clamped(Bb, ldb, brow, N, k0 + kA0, k0 + kB0, K)
                             : ld_col_clamped(Bb, ldb, brow, N, k0 + kA0, k0 + kB0, K);
            acc = __builtin_amdgcn_wmma_f32_16x16x32_f16(false, af, false, bf,
                                                         (short)0, acc, false, false);
        }
    }
    if (K32 < K) {   // K tail (clamped k)
        v16h af = ld_row_clamped(Ab, lda, arow, M, K32 + kA0, K32 + kB0, K);
        v16h bf = bTrans ? ld_row_clamped(Bb, ldb, brow, N, K32 + kA0, K32 + kB0, K)
                         : ld_col_clamped(Bb, ldb, brow, N, K32 + kA0, K32 + kB0, K);
        acc = __builtin_amdgcn_wmma_f32_16x16x32_f16(false, af, false, bf,
                                                     (short)0, acc, false, false);
    }

    int   cn = n0 + r16;
    float bv = (bias != nullptr && cn < N) ? bias[cn] : 0.0f;
#pragma unroll
    for (int r = 0; r < 8; ++r) {
        int cm = m0 + r + 8 * hi;            // C/D layout: M = vgpr + 8*(lane>=16)
        if (cm < M && cn < N) {
            float v = acc[r] * alpha + bv;
            if (act == 1) v = v > 0.0f ? v : 0.0f;
            Cb[(long long)cm * ldc + cn] = v;
        }
    }
}

// ---------------------------------------------------------------------------
// Fused attention: P = softmax(Q K^T * scale) written once to global.
// One 256-thread block (8 waves) computes a full 16-row x L strip.
// Each wave: 8 WMMA accumulators (Q fragment reused), cross-lane max/sum via
// __shfl_xor within 16-lane halves, cross-wave via tiny LDS arrays.
// ---------------------------------------------------------------------------
__global__ void __launch_bounds__(256)
attn_fused(const float* __restrict__ Q, int ldq, long long sQb, long long sQh,
           const float* __restrict__ Kmat, int ldk, long long sKb, long long sKh,
           float* __restrict__ P, long long sPb, long long sPh,
           int M, int L, int nH, float scale)
{
    __shared__ float redm[16][8];
    __shared__ float rowm[16];
    __shared__ float reds[16][8];
    __shared__ float rowsum[16];

    int z = blockIdx.y, bb = z / nH, hh = z % nH;
    const float* Qb = Q    + bb * sQb + hh * sQh;
    const float* Kb = Kmat + bb * sKb + hh * sKh;
    float*       Pb = P    + bb * sPb + hh * sPh;

    int m0   = blockIdx.x * 16;
    int wave = threadIdx.y, lane = threadIdx.x;
    int r16  = lane & 15, hi = lane >> 4;
    int kA0  = 8 * hi, kB0 = 16 + 8 * hi;

    // Q fragment (rows m0..m0+15, K=32), reused for all 8 WMMAs
    v16h qf;
    {
        int  qr = m0 + r16;
        int  rc = qr < M ? qr : M - 1;
        bool rv = qr < M;
        const float* p = Qb + (long long)rc * ldq;
#pragma unroll
        for (int i = 0; i < 8; ++i) {
            qf[i]     = rv ? (_Float16)p[kA0 + i] : (_Float16)0.0f;
            qf[i + 8] = rv ? (_Float16)p[kB0 + i] : (_Float16)0.0f;
        }
    }

    int Lt = (L + 15) >> 4;
    v8f acc[8];
#pragma unroll
    for (int j = 0; j < 8; ++j) {
        v8f a;
#pragma unroll
        for (int r = 0; r < 8; ++r) a[r] = -1e30f;   // sentinel: exp() -> 0
        int t = wave + j * 8;
        if (t < Lt) {                                 // wave-uniform
            int  nr = t * 16 + r16;
            int  rc = nr < L ? nr : L - 1;
            const float* p = Kb + (long long)rc * ldk;
            v16h kf;
#pragma unroll
            for (int i = 0; i < 8; ++i) {
                kf[i]     = (_Float16)p[kA0 + i];
                kf[i + 8] = (_Float16)p[kB0 + i];
            }
            v8f a0 = {};
            a0 = __builtin_amdgcn_wmma_f32_16x16x32_f16(false, qf, false, kf,
                                                        (short)0, a0, false, false);
            if (nr < L) {
#pragma unroll
                for (int r = 0; r < 8; ++r) a[r] = a0[r] * scale;
            }
        }
        acc[j] = a;
    }

    // per-row max: over 8 tiles, then 16 lanes in half-wave, then 8 waves
#pragma unroll
    for (int r = 0; r < 8; ++r) {
        float m = acc[0][r];
#pragma unroll
        for (int j = 1; j < 8; ++j) m = fmaxf(m, acc[j][r]);
#pragma unroll
        for (int s = 1; s < 16; s <<= 1) m = fmaxf(m, __shfl_xor(m, s, 32));
        if (r16 == 0) redm[r + 8 * hi][wave] = m;
    }
    __syncthreads();
    int tid = wave * 32 + lane;
    if (tid < 16) {
        float m = redm[tid][0];
#pragma unroll
        for (int i = 1; i < 8; ++i) m = fmaxf(m, redm[tid][i]);
        rowm[tid] = m;
    }
    __syncthreads();

    // exp + per-row sum
#pragma unroll
    for (int r = 0; r < 8; ++r) {
        float mx = rowm[r + 8 * hi];
        float s = 0.0f;
#pragma unroll
        for (int j = 0; j < 8; ++j) { float e = __expf(acc[j][r] - mx); acc[j][r] = e; s += e; }
#pragma unroll
        for (int sh = 1; sh < 16; sh <<= 1) s += __shfl_xor(s, sh, 32);
        if (r16 == 0) reds[r + 8 * hi][wave] = s;
    }
    __syncthreads();
    if (tid < 16) {
        float s = 0.0f;
#pragma unroll
        for (int i = 0; i < 8; ++i) s += reds[tid][i];
        rowsum[tid] = s;
    }
    __syncthreads();

    float inv[8];
#pragma unroll
    for (int r = 0; r < 8; ++r) inv[r] = 1.0f / rowsum[r + 8 * hi];

    // write normalized probabilities (single HBM pass over P)
#pragma unroll
    for (int j = 0; j < 8; ++j) {
        int t = wave + j * 8;
        if (t >= Lt) continue;
        int n = t * 16 + r16;
        if (n >= L) continue;
#pragma unroll
        for (int r = 0; r < 8; ++r) {
            int m = m0 + r + 8 * hi;
            if (m < M) Pb[(long long)m * L + n] = acc[j][r] * inv[r];
        }
    }
}

// ------------------------------- helpers -----------------------------------
__global__ void fill_f32(float* p, float v, long long n) {
    long long i = (long long)blockIdx.x * 256 + threadIdx.x;
    if (i < n) p[i] = v;
}
__global__ void transpose_kn(const float* __restrict__ in, float* __restrict__ outp,
                             int Kd, int Nd) {   // (K,N) -> (N,K)
    long long i = (long long)blockIdx.x * 256 + threadIdx.x;
    if (i < (long long)Kd * Nd) {
        int k = (int)(i / Nd), n = (int)(i % Nd);
        outp[(long long)n * Kd + k] = in[i];
    }
}
__global__ void deg_accum(const int* __restrict__ dst, const float* __restrict__ ew,
                          float* __restrict__ deg, int E) {
    int e = blockIdx.x * 256 + threadIdx.x;
    if (e < E) atomicAdd(&deg[dst[e]], ew[e]);
}
__global__ void rsqrt_inplace(float* p, int n) {
    int i = blockIdx.x * 256 + threadIdx.x;
    if (i < n) p[i] = rsqrtf(p[i]);
}
__global__ void agg_init(const float* __restrict__ h, const float* __restrict__ dinv,
                         const float* __restrict__ bias, float* __restrict__ out) {
    long long v = blockIdx.x;
    int c = threadIdx.x;
    float di = dinv[v];
    out[v * HBLK + c] = bias[c] + di * di * h[v * HBLK + c];
}
__global__ void agg_edges(const int* __restrict__ src, const int* __restrict__ dst,
                          const float* __restrict__ ew, const float* __restrict__ dinv,
                          const float* __restrict__ h, float* __restrict__ out) {
    long long e = blockIdx.x;
    int c = threadIdx.x;
    int s = src[e], d = dst[e];
    float coeff = dinv[s] * ew[e] * dinv[d];
    atomicAdd(&out[(long long)d * HBLK + c], coeff * h[(long long)s * HBLK + c]);
}
__global__ void relu_inplace(float* p, long long n) {
    long long i = (long long)blockIdx.x * 256 + threadIdx.x;
    if (i < n) p[i] = p[i] > 0.0f ? p[i] : 0.0f;
}
__global__ void pool_scatter(const float* __restrict__ h, const int* __restrict__ roi,
                             const int* __restrict__ batch, float* __restrict__ sum,
                             float* __restrict__ cnt, int nroi) {
    long long v = blockIdx.x;
    int c = threadIdx.x;
    int key = batch[v] * nroi + roi[v];
    atomicAdd(&sum[(long long)key * HBLK + c], h[v * HBLK + c]);
    if (c == 0) atomicAdd(&cnt[key], 1.0f);
}
__global__ void pool_div(float* __restrict__ sum, const float* __restrict__ cnt) {
    long long k = blockIdx.x;
    int c = threadIdx.x;
    sum[k * HBLK + c] /= fmaxf(cnt[k], 1.0f);
}
__global__ void add2(const float* __restrict__ a, const float* __restrict__ b,
                     float* __restrict__ c, long long n) {
    long long i = (long long)blockIdx.x * 256 + threadIdx.x;
    if (i < n) c[i] = a[i] + b[i];
}
__global__ void add_layernorm(const float* __restrict__ X, const float* __restrict__ A,
                              const float* __restrict__ g, const float* __restrict__ b,
                              float* __restrict__ Y) {
    long long row = blockIdx.x;
    int t = threadIdx.x;
    float v = X[row * HBLK + t] + A[row * HBLK + t];
    __shared__ float red[HBLK];
    red[t] = v; __syncthreads();
    for (int s = 64; s > 0; s >>= 1) { if (t < s) red[t] += red[t + s]; __syncthreads(); }
    float mu = red[0] * (1.0f / HBLK);
    __syncthreads();
    float d = v - mu;
    red[t] = d * d; __syncthreads();
    for (int s = 64; s > 0; s >>= 1) { if (t < s) red[t] += red[t + s]; __syncthreads(); }
    float var = red[0] * (1.0f / HBLK);
    Y[row * HBLK + t] = d * rsqrtf(var + 1e-5f) * g[t] + b[t];
}
__global__ void bn_leaky(float* __restrict__ p, const float* __restrict__ g,
                         const float* __restrict__ b, int N1, long long total) {
    long long i = (long long)blockIdx.x * 256 + threadIdx.x;
    if (i < total) {
        int c = (int)(i % N1);
        float v = p[i] * (g[c] * rsqrtf(1.0f + 1e-5f)) + b[c];
        p[i] = v > 0.0f ? v : 0.01f * v;
    }
}

// ------------------------------ host side ----------------------------------
static inline void launch_gemm(hipStream_t s,
    const float* A, int lda, long long sAb, long long sAh,
    const float* Bm, int ldb, long long sBb, long long sBh,
    float* C, int ldc, long long sCb, long long sCh,
    const float* bias, int M, int N, int K, int nB, int nH,
    int bTrans, float alpha, int act)
{
    int mt = (M + 15) / 16, nt = (N + 15) / 16;
    dim3 grid((nt + 3) / 4, mt, nB * nH);
    dim3 block(32, 4, 1);
    gemm_wmma_f16<<<grid, block, 0, s>>>(A, lda, sAb, sAh, Bm, ldb, sBb, sBh,
                                         C, ldc, sCb, sCh, bias,
                                         M, N, K, nH, bTrans, alpha, act);
}

extern "C" void kernel_launch(void* const* d_in, const int* in_sizes, int n_in,
                              void* d_out, int out_size, void* d_ws, size_t ws_size,
                              hipStream_t stream) {
    const int Bg = 16, NPER = 1024, NV = 16384, E = 262144;
    const int NROI = 148, N2 = 2368, E2 = 65536, DIN = 64, H = 128, NH = 4;
    const float SC = 0.17677669529663687f;   // 1/sqrt(32)

    const float* x      = (const float*)d_in[0];
    const int*   nroi   = (const int*)  d_in[1];
    const int*   batc   = (const int*)  d_in[2];
    const int*   ei     = (const int*)  d_in[3];   // src=ei[0:E], dst=ei[E:2E]
    const float* ew     = (const float*)d_in[4];
    const float* x2     = (const float*)d_in[5];
    const int*   roi2   = (const int*)  d_in[6];
    const int*   batc2  = (const int*)  d_in[7];
    const int*   ei2    = (const int*)  d_in[8];
    const float* ew2    = (const float*)d_in[9];
    const float* gW1 = (const float*)d_in[10], *gb1 = (const float*)d_in[11];
    const float* gW2 = (const float*)d_in[12], *gb2 = (const float*)d_in[13];
    const float* rW1 = (const float*)d_in[14], *rb1 = (const float*)d_in[15];
    const float* rW2 = (const float*)d_in[16], *rb2 = (const float*)d_in[17];
    const float* m_inw = (const float*)d_in[18], *m_inb = (const float*)d_in[19];
    const float* a_inw  = (const float*)d_in[30], *a_inb  = (const float*)d_in[31];
    const float* a_outw = (const float*)d_in[32], *a_outb = (const float*)d_in[33];
    const float* a_ln1g = (const float*)d_in[34], *a_ln1b = (const float*)d_in[35];
    const float* a_ff1w = (const float*)d_in[36], *a_ff1b = (const float*)d_in[37];
    const float* a_ff2w = (const float*)d_in[38], *a_ff2b = (const float*)d_in[39];
    const float* a_ln2g = (const float*)d_in[40], *a_ln2b = (const float*)d_in[41];
    const float* w1  = (const float*)d_in[42], *b1  = (const float*)d_in[43];
    const float* bng = (const float*)d_in[44], *bnb = (const float*)d_in[45];
    const float* w2  = (const float*)d_in[46], *b2  = (const float*)d_in[47];

    float* out = (float*)d_out;
    const long long O_OUT = 0, O_EMB = 32, O_EMBROI = 303136, O_COMB = 606240,
                    O_TOUT = 909344, O_ATTNW = 1212448, O_E3H = 2614304,
                    O_W3H = 4711456;

    float* ws = (float*)d_ws;
    size_t off = 0;
    auto arena = [&](size_t n) { size_t o = off; off += (n + 63) & ~(size_t)63; return o; };
    float* deg1   = ws + arena(NV);
    float* deg2   = ws + arena(N2);
    float* cnt    = ws + arena(Bg * NROI);
    float* hbuf   = ws + arena((size_t)NV * H);
    float* aggbuf = ws + arena((size_t)NV * H);
    float* qkvbig = ws + arena((size_t)NV * 256);
    float* hroiA  = ws + arena((size_t)N2 * H);
    float* hroiB  = ws + arena((size_t)N2 * H);
    float* qkv2   = ws + arena((size_t)N2 * 384);
    float* atto   = ws + arena((size_t)N2 * H);
    float* abuf   = ws + arena((size_t)N2 * H);
    float* x1b    = ws + arena((size_t)N2 * H);
    float* t1b    = ws + arena((size_t)N2 * H);
    float* t2b    = ws + arena((size_t)N2 * H);
    float* hcb    = ws + arena((size_t)Bg * 1000);
    float* wtg1   = ws + arena((size_t)H * DIN);  // gW1^T (128,64)
    float* wtg2   = ws + arena((size_t)H * H);    // gW2^T (128,128)
    float* wtr1   = ws + arena((size_t)H * DIN);
    float* wtr2   = ws + arena((size_t)H * H);
    (void)ws_size; (void)in_sizes; (void)n_in; (void)out_size;

    auto g256 = [](long long n) { return dim3((unsigned)((n + 255) / 256)); };

    // pre-transpose GCN weights -> every big GEMM takes the vectorized path
    transpose_kn<<<g256(DIN * H), 256, 0, stream>>>(gW1, wtg1, DIN, H);
    transpose_kn<<<g256(H * H),   256, 0, stream>>>(gW2, wtg2, H, H);
    transpose_kn<<<g256(DIN * H), 256, 0, stream>>>(rW1, wtr1, DIN, H);
    transpose_kn<<<g256(H * H),   256, 0, stream>>>(rW2, wtr2, H, H);

    // ================= ROI-graph GCN stack =================
    fill_f32<<<g256(N2), 256, 0, stream>>>(deg2, 1.0f, N2);
    deg_accum<<<g256(E2), 256, 0, stream>>>(ei2 + E2, ew2, deg2, E2);
    rsqrt_inplace<<<g256(N2), 256, 0, stream>>>(deg2, N2);
    launch_gemm(stream, x2, DIN, 0, 0, wtr1, DIN, 0, 0, hroiA, H, 0, 0,
                nullptr, N2, H, DIN, 1, 1, 1, 1.0f, 0);
    agg_init<<<N2, H, 0, stream>>>(hroiA, deg2, rb1, hroiB);
    agg_edges<<<E2, H, 0, stream>>>(ei2, ei2 + E2, ew2, deg2, hroiA, hroiB);
    relu_inplace<<<g256((long long)N2 * H), 256, 0, stream>>>(hroiB, (long long)N2 * H);
    launch_gemm(stream, hroiB, H, 0, 0, wtr2, H, 0, 0, hroiA, H, 0, 0,
                nullptr, N2, H, H, 1, 1, 1, 1.0f, 0);
    agg_init<<<N2, H, 0, stream>>>(hroiA, deg2, rb2, hroiB);
    agg_edges<<<E2, H, 0, stream>>>(ei2, ei2 + E2, ew2, deg2, hroiA, hroiB);
    relu_inplace<<<g256((long long)N2 * H), 256, 0, stream>>>(hroiB, (long long)N2 * H);

    // ================= vertex-graph GCN stack =================
    fill_f32<<<g256(NV), 256, 0, stream>>>(deg1, 1.0f, NV);
    deg_accum<<<g256(E), 256, 0, stream>>>(ei + E, ew, deg1, E);
    rsqrt_inplace<<<g256(NV), 256, 0, stream>>>(deg1, NV);
    launch_gemm(stream, x, DIN, 0, 0, wtg1, DIN, 0, 0, hbuf, H, 0, 0,
                nullptr, NV, H, DIN, 1, 1, 1, 1.0f, 0);
    agg_init<<<NV, H, 0, stream>>>(hbuf, deg1, gb1, aggbuf);
    agg_edges<<<E, H, 0, stream>>>(ei, ei + E, ew, deg1, hbuf, aggbuf);
    relu_inplace<<<g256((long long)NV * H), 256, 0, stream>>>(aggbuf, (long long)NV * H);
    launch_gemm(stream, aggbuf, H, 0, 0, wtg2, H, 0, 0, hbuf, H, 0, 0,
                nullptr, NV, H, H, 1, 1, 1, 1.0f, 0);
    agg_init<<<NV, H, 0, stream>>>(hbuf, deg1, gb2, out + O_E3H);
    agg_edges<<<E, H, 0, stream>>>(ei, ei + E, ew, deg1, hbuf, out + O_E3H);
    relu_inplace<<<g256((long long)NV * H), 256, 0, stream>>>(out + O_E3H, (long long)NV * H);

    // ================= ROI mean-pools + combined =================
    fill_f32<<<g256((long long)Bg * NROI * H), 256, 0, stream>>>(out + O_EMB, 0.0f, (long long)Bg * NROI * H);
    fill_f32<<<g256(Bg * NROI), 256, 0, stream>>>(cnt, 0.0f, Bg * NROI);
    pool_scatter<<<NV, H, 0, stream>>>(out + O_E3H, nroi, batc, out + O_EMB, cnt, NROI);
    pool_div<<<Bg * NROI, H, 0, stream>>>(out + O_EMB, cnt);

    fill_f32<<<g256((long long)Bg * NROI * H), 256, 0, stream>>>(out + O_EMBROI, 0.0f, (long long)Bg * NROI * H);
    fill_f32<<<g256(Bg * NROI), 256, 0, stream>>>(cnt, 0.0f, Bg * NROI);
    pool_scatter<<<N2, H, 0, stream>>>(hroiB, roi2, batc2, out + O_EMBROI, cnt, NROI);
    pool_div<<<Bg * NROI, H, 0, stream>>>(out + O_EMBROI, cnt);

    add2<<<g256((long long)Bg * NROI * H), 256, 0, stream>>>(out + O_EMB, out + O_EMBROI,
                                                             out + O_COMB, (long long)Bg * NROI * H);

    // ========== big attention: w3h = softmax(QK^T/sqrt(32)), fused ==========
    launch_gemm(stream, out + O_E3H, H, 0, 0, m_inw, H, 0, 0, qkvbig, 256, 0, 0,
                m_inb, NV, 256, H, 1, 1, 1, 1.0f, 0);
    attn_fused<<<dim3(NPER / 16, Bg * NH), dim3(32, 8), 0, stream>>>(
        qkvbig,       256, (long long)NPER * 256, 32,
        qkvbig + 128, 256, (long long)NPER * 256, 32,
        out + O_W3H, (long long)NH * NPER * NPER, (long long)NPER * NPER,
        NPER, NPER, NH, SC);

    // ================= small attention block on combined =================
    launch_gemm(stream, out + O_COMB, H, 0, 0, a_inw, H, 0, 0, qkv2, 384, 0, 0,
                a_inb, N2, 384, H, 1, 1, 1, 1.0f, 0);
    attn_fused<<<dim3((NROI + 15) / 16, Bg * NH), dim3(32, 8), 0, stream>>>(
        qkv2,       384, (long long)NROI * 384, 32,
        qkv2 + 128, 384, (long long)NROI * 384, 32,
        out + O_ATTNW, (long long)NH * NROI * NROI, (long long)NROI * NROI,
        NROI, NROI, NH, SC);
    // P @ V -> atto (B*148,128) head-interleaved
    launch_gemm(stream,
                out + O_ATTNW, NROI, (long long)NH * NROI * NROI, (long long)NROI * NROI,
                qkv2 + 256,    384,  (long long)NROI * 384, 32,
                atto, H, (long long)NROI * H, 32,
                nullptr, NROI, 32, NROI, Bg, NH, 0, 1.0f, 0);
    launch_gemm(stream, atto, H, 0, 0, a_outw, H, 0, 0, abuf, H, 0, 0,
                a_outb, N2, H, H, 1, 1, 1, 1.0f, 0);
    add_layernorm<<<N2, H, 0, stream>>>(out + O_COMB, abuf, a_ln1g, a_ln1b, x1b);
    launch_gemm(stream, x1b, H, 0, 0, a_ff1w, H, 0, 0, t1b, H, 0, 0,
                a_ff1b, N2, H, H, 1, 1, 1, 1.0f, 1);
    launch_gemm(stream, t1b, H, 0, 0, a_ff2w, H, 0, 0, t2b, H, 0, 0,
                a_ff2b, N2, H, H, 1, 1, 1, 1.0f, 0);
    add_layernorm<<<N2, H, 0, stream>>>(x1b, t2b, a_ln2g, a_ln2b, out + O_TOUT);

    // ================= classifier MLP =================
    const int KF = NROI * H;  // 18944
    launch_gemm(stream, out + O_TOUT, KF, 0, 0, w1, KF, 0, 0, hcb, 1000, 0, 0,
                b1, Bg, 1000, KF, 1, 1, 1, 1.0f, 0);
    bn_leaky<<<g256((long long)Bg * 1000), 256, 0, stream>>>(hcb, bng, bnb, 1000, (long long)Bg * 1000);
    launch_gemm(stream, hcb, 1000, 0, 0, w2, 1000, 0, 0, out + O_OUT, 2, 0, 0,
                b2, Bg, 2, 1000, 1, 1, 1, 1.0f, 0);
}
